// SequentialLlama4TextMoe_24696061952101
// MI455X (gfx1250) — matrix operand order repaired and checked by reading, not compiled, not run on hardware
//
#include <hip/hip_runtime.h>
#include <hip/hip_bf16.h>

typedef __attribute__((ext_vector_type(16))) _Float16 v16h;
typedef __attribute__((ext_vector_type(8)))  _Float16 v8h;
typedef __attribute__((ext_vector_type(4)))  _Float16 v4h;
typedef __attribute__((ext_vector_type(8)))  float    v8f;

#define E_ 8
#define H_ 2048
#define I_ 4096
#define T_ 2048

// workspace layout (bytes)
#define WS_HDR    0                   // 24 ints: counts[8], offsets[8], cursors[8]
#define WS_EXPERT 256                 // int[T]
#define WS_SCORE  (256 + 4*T_)        // float[T]
#define WS_PERM   (256 + 8*T_)        // int[T]
#define WS_SACT   32768               // _Float16[T*I]
#define WS_AACT   (32768 + 2*(size_t)T_*I_)

// fast sigmoid: v_exp_f32 + v_rcp_f32 (f16-storage precision; avoids IEEE div chain)
__device__ __forceinline__ float sigmoidf_(float x) {
  return __builtin_amdgcn_rcpf(1.0f + __expf(-x));
}

// ---------------------------------------------------------------- router ----
__global__ void moe_init(int* hdr) {
  if (threadIdx.x < 24) hdr[threadIdx.x] = 0;
}

__global__ __launch_bounds__(256)
void moe_router(const float* __restrict__ x, const float* __restrict__ rw,
                float* __restrict__ scores_out, int* __restrict__ expert,
                float* __restrict__ score, int* __restrict__ hdr) {
  const int t = blockIdx.x;
  const int tid = threadIdx.x;
  float acc[E_];
#pragma unroll
  for (int e = 0; e < E_; ++e) acc[e] = 0.f;
  const float* xr = x + (size_t)t * H_;
  for (int h = tid; h < H_; h += 256) {
    float xv = xr[h];
#pragma unroll
    for (int e = 0; e < E_; ++e) acc[e] += xv * rw[e * H_ + h];
  }
#pragma unroll
  for (int e = 0; e < E_; ++e) {
#pragma unroll
    for (int off = 16; off > 0; off >>= 1) acc[e] += __shfl_down(acc[e], off, 32);
  }
  __shared__ float red[8][E_];
  const int wave = tid >> 5, lane = tid & 31;
  if (lane == 0) {
#pragma unroll
    for (int e = 0; e < E_; ++e) red[wave][e] = acc[e];
  }
  __syncthreads();
  if (tid == 0) {
    float best = -3.0e38f; int be = 0;
#pragma unroll
    for (int e = 0; e < E_; ++e) {
      float v = 0.f;
#pragma unroll
      for (int w = 0; w < 8; ++w) v += red[w][e];
      if (v > best) { best = v; be = e; }
    }
    float s = sigmoidf_(best);
#pragma unroll
    for (int e = 0; e < E_; ++e)
      scores_out[(size_t)e * T_ + t] = (e == be) ? s : 0.f;
    expert[t] = be;
    score[t] = s;
    atomicAdd(&hdr[be], 1);
  }
}

__global__ void moe_scan(int* hdr) {
  if (threadIdx.x == 0) {
    int o = 0;
    for (int e = 0; e < E_; ++e) { hdr[8 + e] = o; o += hdr[e]; }
  }
}

__global__ void moe_assign(const int* __restrict__ expert, int* hdr, int* __restrict__ perm) {
  int t = blockIdx.x * 256 + threadIdx.x;
  int e = expert[t];
  int pos = atomicAdd(&hdr[16 + e], 1);
  perm[hdr[8 + e] + pos] = t;
}

// helper: stage a 4x4 f32 micro-block as 4 k-contiguous v4h columns (ds_store_b64)
__device__ __forceinline__ void stage_b4x4(_Float16 (*Bs)[40], int nq, int kg,
                                           const float4& r0, const float4& r1,
                                           const float4& r2, const float4& r3) {
  v4h h;
  h[0]=(_Float16)r0.x; h[1]=(_Float16)r1.x; h[2]=(_Float16)r2.x; h[3]=(_Float16)r3.x;
  *(v4h*)&Bs[nq + 0][kg] = h;
  h[0]=(_Float16)r0.y; h[1]=(_Float16)r1.y; h[2]=(_Float16)r2.y; h[3]=(_Float16)r3.y;
  *(v4h*)&Bs[nq + 1][kg] = h;
  h[0]=(_Float16)r0.z; h[1]=(_Float16)r1.z; h[2]=(_Float16)r2.z; h[3]=(_Float16)r3.z;
  *(v4h*)&Bs[nq + 2][kg] = h;
  h[0]=(_Float16)r0.w; h[1]=(_Float16)r1.w; h[2]=(_Float16)r2.w; h[3]=(_Float16)r3.w;
  *(v4h*)&Bs[nq + 3][kg] = h;
}

// -------------------------------------------------- fused gate/up (shared) --
__global__ __launch_bounds__(256)
void gemm_gateup_shared(const float* __restrict__ X, const float* __restrict__ Wg,
                        const float* __restrict__ Wu, _Float16* __restrict__ Sact) {
  __shared__ _Float16 As[128][40];
  __shared__ _Float16 Bgs[64][40];
  __shared__ _Float16 Bus[64][40];
  const int tid = threadIdx.x;
  const int lane = tid & 31, wave = tid >> 5;
  const int half = lane >> 4, lr = lane & 15;
  const int mWave = (wave >> 2) * 64, nWave = (wave & 3) * 16;
  const int mBase = blockIdx.y * 128, nBase = blockIdx.x * 64;

  // B staging role: threads 0-127 gate, 128-255 up; 4x4 micro-block each
  const int bsel = tid >> 7;
  const int bnq  = (tid & 15) * 4;
  const int bkg  = ((tid >> 4) & 7) * 4;
  const float* Wsel = bsel ? Wu : Wg;
  _Float16 (*Bsel)[40] = bsel ? Bus : Bgs;

  v8f accG[4], accU[4];
#pragma unroll
  for (int mi = 0; mi < 4; ++mi) {
    accG[mi] = (v8f){0.f,0.f,0.f,0.f,0.f,0.f,0.f,0.f};
    accU[mi] = (v8f){0.f,0.f,0.f,0.f,0.f,0.f,0.f,0.f};
  }

  for (int k0 = 0; k0 < H_; k0 += 32) {
    __syncthreads();
#pragma unroll
    for (int i = 0; i < 4; ++i) {          // A: 128x32 fp32 -> f16 (v4h = ds_store_b64)
      int q = tid + 256 * i;
      int r = q >> 3, kq = (q & 7) * 4;
      float4 v = *(const float4*)(X + (size_t)(mBase + r) * H_ + k0 + kq);
      v4h h; h[0] = (_Float16)v.x; h[1] = (_Float16)v.y; h[2] = (_Float16)v.z; h[3] = (_Float16)v.w;
      *(v4h*)&As[r][kq] = h;
    }
    {                                       // B: 32x64 each, col-major LDS, b64 stores
      const float* src = Wsel + (size_t)(k0 + bkg) * I_ + nBase + bnq;
      float4 r0 = *(const float4*)(src);
      float4 r1 = *(const float4*)(src + I_);
      float4 r2 = *(const float4*)(src + 2 * I_);
      float4 r3 = *(const float4*)(src + 3 * I_);
      stage_b4x4(Bsel, bnq, bkg, r0, r1, r2, r3);
      if (k0 + 32 < H_) __builtin_prefetch(src + 32 * I_, 0, 0);  // global_prefetch_b8
    }
    __syncthreads();

    v16h a[4];
#pragma unroll
    for (int mi = 0; mi < 4; ++mi) {
      v8h lo = *(const v8h*)&As[mWave + mi * 16 + lr][half * 8];
      v8h hi = *(const v8h*)&As[mWave + mi * 16 + lr][16 + half * 8];
#pragma unroll
      for (int j = 0; j < 8; ++j) { a[mi][j] = lo[j]; a[mi][8 + j] = hi[j]; }
    }
    v16h bg, bu;
    {
      v8h l0 = *(const v8h*)&Bgs[nWave + lr][half * 8];
      v8h l1 = *(const v8h*)&Bgs[nWave + lr][16 + half * 8];
      v8h m0 = *(const v8h*)&Bus[nWave + lr][half * 8];
      v8h m1 = *(const v8h*)&Bus[nWave + lr][16 + half * 8];
#pragma unroll
      for (int j = 0; j < 8; ++j) { bg[j] = l0[j]; bg[8 + j] = l1[j]; bu[j] = m0[j]; bu[8 + j] = m1[j]; }
    }
#pragma unroll
    for (int mi = 0; mi < 4; ++mi) {
      accG[mi] = __builtin_amdgcn_wmma_f32_16x16x32_f16(false, a[mi], false, bg, (short)0, accG[mi], false, false);
      accU[mi] = __builtin_amdgcn_wmma_f32_16x16x32_f16(false, a[mi], false, bu, (short)0, accU[mi], false, false);
    }
  }

  const int col = nBase + nWave + lr;
#pragma unroll
  for (int mi = 0; mi < 4; ++mi) {
    int rowb = mBase + mWave + mi * 16 + half * 8;
#pragma unroll
    for (int r = 0; r < 8; ++r) {
      float g = accG[mi][r], u = accU[mi][r];
      float act = g * sigmoidf_(g) * u;   // silu(gate)*up
      Sact[(size_t)(rowb + r) * I_ + col] = (_Float16)act;
    }
  }
}

// -------------------------------------------------- fused gate/up (routed) --
__global__ __launch_bounds__(256)
void gemm_gateup_routed(const float* __restrict__ X, const float* __restrict__ Wgu,
                        _Float16* __restrict__ Aact, const int* __restrict__ hdr,
                        const int* __restrict__ perm, const float* __restrict__ score) {
  const int e = blockIdx.z;
  const int cnt = hdr[e];
  const int off = hdr[8 + e];
  if ((int)blockIdx.y * 128 >= cnt) return;

  __shared__ _Float16 As[128][40];
  __shared__ _Float16 Bgs[64][40];
  __shared__ _Float16 Bus[64][40];
  __shared__ int   rowTok[128];
  __shared__ float rowScale[128];

  const int tid = threadIdx.x;
  const int lane = tid & 31, wave = tid >> 5;
  const int half = lane >> 4, lr = lane & 15;
  const int mWave = (wave >> 2) * 64, nWave = (wave & 3) * 16;
  const int mBase = blockIdx.y * 128, nBase = blockIdx.x * 64;
  const float* Wb = Wgu + (size_t)e * H_ * (2 * I_);

  const int bsel = tid >> 7;            // 0 -> gate cols, 1 -> up cols (offset I_)
  const int bnq  = (tid & 15) * 4;
  const int bkg  = ((tid >> 4) & 7) * 4;
  const float* Wcol = Wb + (bsel ? I_ : 0) + nBase + bnq;
  _Float16 (*Bsel)[40] = bsel ? Bus : Bgs;

  if (tid < 128) {
    int r = mBase + tid;
    int tok = (r < cnt) ? perm[off + r] : 0;
    rowTok[tid] = tok;
    rowScale[tid] = (r < cnt) ? score[tok] : 0.f;   // zero rows for padding
  }
  __syncthreads();

  v8f accG[4], accU[4];
#pragma unroll
  for (int mi = 0; mi < 4; ++mi) {
    accG[mi] = (v8f){0.f,0.f,0.f,0.f,0.f,0.f,0.f,0.f};
    accU[mi] = (v8f){0.f,0.f,0.f,0.f,0.f,0.f,0.f,0.f};
  }

  for (int k0 = 0; k0 < H_; k0 += 32) {
    __syncthreads();
#pragma unroll
    for (int i = 0; i < 4; ++i) {          // A: gathered, scaled rows
      int q = tid + 256 * i;
      int r = q >> 3, kq = (q & 7) * 4;
      float s = rowScale[r];
      float4 v = *(const float4*)(X + (size_t)rowTok[r] * H_ + k0 + kq);
      v4h h; h[0] = (_Float16)(v.x * s); h[1] = (_Float16)(v.y * s);
             h[2] = (_Float16)(v.z * s); h[3] = (_Float16)(v.w * s);
      *(v4h*)&As[r][kq] = h;
    }
    {                                       // B: 32x64 (gate or up), b64 stores
      const float* src = Wcol + (size_t)(k0 + bkg) * (2 * I_);
      float4 r0 = *(const float4*)(src);
      float4 r1 = *(const float4*)(src + 2 * I_);
      float4 r2 = *(const float4*)(src + 4 * I_);
      float4 r3 = *(const float4*)(src + 6 * I_);
      stage_b4x4(Bsel, bnq, bkg, r0, r1, r2, r3);
      if (k0 + 32 < H_) __builtin_prefetch(src + (size_t)32 * 2 * I_, 0, 0);
    }
    __syncthreads();

    v16h a[4];
#pragma unroll
    for (int mi = 0; mi < 4; ++mi) {
      v8h lo = *(const v8h*)&As[mWave + mi * 16 + lr][half * 8];
      v8h hi = *(const v8h*)&As[mWave + mi * 16 + lr][16 + half * 8];
#pragma unroll
      for (int j = 0; j < 8; ++j) { a[mi][j] = lo[j]; a[mi][8 + j] = hi[j]; }
    }
    v16h bg, bu;
    {
      v8h l0 = *(const v8h*)&Bgs[nWave + lr][half * 8];
      v8h l1 = *(const v8h*)&Bgs[nWave + lr][16 + half * 8];
      v8h m0 = *(const v8h*)&Bus[nWave + lr][half * 8];
      v8h m1 = *(const v8h*)&Bus[nWave + lr][16 + half * 8];
#pragma unroll
      for (int j = 0; j < 8; ++j) { bg[j] = l0[j]; bg[8 + j] = l1[j]; bu[j] = m0[j]; bu[8 + j] = m1[j]; }
    }
#pragma unroll
    for (int mi = 0; mi < 4; ++mi) {
      accG[mi] = __builtin_amdgcn_wmma_f32_16x16x32_f16(false, a[mi], false, bg, (short)0, accG[mi], false, false);
      accU[mi] = __builtin_amdgcn_wmma_f32_16x16x32_f16(false, a[mi], false, bu, (short)0, accU[mi], false, false);
    }
  }

  const int col = nBase + nWave + lr;
#pragma unroll
  for (int mi = 0; mi < 4; ++mi) {
    int lrow0 = mWave + mi * 16 + half * 8;
#pragma unroll
    for (int r = 0; r < 8; ++r) {
      int lrow = mBase + lrow0 + r;
      if (lrow < cnt) {
        float g = accG[mi][r], u = accU[mi][r];
        float act = g * sigmoidf_(g) * u;
        Aact[(size_t)(off + lrow) * I_ + col] = (_Float16)act;
      }
    }
  }
}

// ------------------------------------------------------- down (shared) -----
__global__ __launch_bounds__(256)
void gemm_down_shared(const _Float16* __restrict__ Sact, const float* __restrict__ Wd,
                      float* __restrict__ out) {
  __shared__ _Float16 As[128][40];
  __shared__ _Float16 Bs[128][40];
  const int tid = threadIdx.x;
  const int lane = tid & 31, wave = tid >> 5;
  const int half = lane >> 4, lr = lane & 15;
  const int mWave = (wave >> 2) * 64, nWave = (wave & 3) * 32;
  const int mBase = blockIdx.y * 128, nBase = blockIdx.x * 128;

  const int bnq = (tid & 31) * 4;       // n within 128
  const int bkg = (tid >> 5) * 4;       // k within 32

  v8f acc[4][2];
#pragma unroll
  for (int mi = 0; mi < 4; ++mi)
#pragma unroll
    for (int ni = 0; ni < 2; ++ni)
      acc[mi][ni] = (v8f){0.f,0.f,0.f,0.f,0.f,0.f,0.f,0.f};

  for (int k0 = 0; k0 < I_; k0 += 32) {
    __syncthreads();
#pragma unroll
    for (int i = 0; i < 2; ++i) {          // A: 128x32 f16 straight copy (b128)
      int q = tid + 256 * i;
      int r = q >> 2, kq = (q & 3) * 8;
      v8h h = *(const v8h*)(Sact + (size_t)(mBase + r) * I_ + k0 + kq);
      *(v8h*)&As[r][kq] = h;
    }
    {                                       // B: 32x128 fp32 -> col-major f16, b64 stores
      const float* src = Wd + (size_t)(k0 + bkg) * H_ + nBase + bnq;
      float4 r0 = *(const float4*)(src);
      float4 r1 = *(const float4*)(src + H_);
      float4 r2 = *(const float4*)(src + 2 * H_);
      float4 r3 = *(const float4*)(src + 3 * H_);
      stage_b4x4(Bs, bnq, bkg, r0, r1, r2, r3);
      if (k0 + 32 < I_) __builtin_prefetch(src + 32 * H_, 0, 0);
    }
    __syncthreads();

    v16h a[4], b[2];
#pragma unroll
    for (int mi = 0; mi < 4; ++mi) {
      v8h lo = *(const v8h*)&As[mWave + mi * 16 + lr][half * 8];
      v8h hi = *(const v8h*)&As[mWave + mi * 16 + lr][16 + half * 8];
#pragma unroll
      for (int j = 0; j < 8; ++j) { a[mi][j] = lo[j]; a[mi][8 + j] = hi[j]; }
    }
#pragma unroll
    for (int ni = 0; ni < 2; ++ni) {
      v8h l0 = *(const v8h*)&Bs[nWave + ni * 16 + lr][half * 8];
      v8h l1 = *(const v8h*)&Bs[nWave + ni * 16 + lr][16 + half * 8];
#pragma unroll
      for (int j = 0; j < 8; ++j) { b[ni][j] = l0[j]; b[ni][8 + j] = l1[j]; }
    }
#pragma unroll
    for (int mi = 0; mi < 4; ++mi)
#pragma unroll
      for (int ni = 0; ni < 2; ++ni)
        acc[mi][ni] = __builtin_amdgcn_wmma_f32_16x16x32_f16(false, a[mi], false, b[ni], (short)0, acc[mi][ni], false, false);
  }

#pragma unroll
  for (int mi = 0; mi < 4; ++mi) {
    int rowb = mBase + mWave + mi * 16 + half * 8;
#pragma unroll
    for (int ni = 0; ni < 2; ++ni) {
      int col = nBase + nWave + ni * 16 + lr;
#pragma unroll
      for (int r = 0; r < 8; ++r)
        out[(size_t)(rowb + r) * H_ + col] = acc[mi][ni][r];
    }
  }
}

// ------------------------------------------------------- down (routed) -----
__global__ __launch_bounds__(256)
void gemm_down_routed(const _Float16* __restrict__ Aact, const float* __restrict__ Wd,
                      float* __restrict__ out, const int* __restrict__ hdr,
                      const int* __restrict__ perm) {
  const int e = blockIdx.z;
  const int cnt = hdr[e];
  const int off = hdr[8 + e];
  if ((int)blockIdx.y * 128 >= cnt) return;

  __shared__ _Float16 As[128][40];
  __shared__ _Float16 Bs[128][40];
  __shared__ int rowTok[128];
  const int tid = threadIdx.x;
  const int lane = tid & 31, wave = tid >> 5;
  const int half = lane >> 4, lr = lane & 15;
  const int mWave = (wave >> 2) * 64, nWave = (wave & 3) * 32;
  const int mBase = blockIdx.y * 128, nBase = blockIdx.x * 128;
  const float* Wb = Wd + (size_t)e * I_ * H_;

  const int bnq = (tid & 31) * 4;
  const int bkg = (tid >> 5) * 4;

  if (tid < 128) {
    int r = mBase + tid;
    rowTok[tid] = (r < cnt) ? perm[off + r] : 0;
  }
  __syncthreads();

  v8f acc[4][2];
#pragma unroll
  for (int mi = 0; mi < 4; ++mi)
#pragma unroll
    for (int ni = 0; ni < 2; ++ni)
      acc[mi][ni] = (v8f){0.f,0.f,0.f,0.f,0.f,0.f,0.f,0.f};

  for (int k0 = 0; k0 < I_; k0 += 32) {
    __syncthreads();
#pragma unroll
    for (int i = 0; i < 2; ++i) {
      int q = tid + 256 * i;
      int r = q >> 2, kq = (q & 3) * 8;
      v8h h = (v8h){0,0,0,0,0,0,0,0};
      if (mBase + r < cnt)
        h = *(const v8h*)(Aact + (size_t)(off + mBase + r) * I_ + k0 + kq);
      *(v8h*)&As[r][kq] = h;
    }
    {
      const float* src = Wb + (size_t)(k0 + bkg) * H_ + nBase + bnq;
      float4 r0 = *(const float4*)(src);
      float4 r1 = *(const float4*)(src + H_);
      float4 r2 = *(const float4*)(src + 2 * H_);
      float4 r3 = *(const float4*)(src + 3 * H_);
      stage_b4x4(Bs, bnq, bkg, r0, r1, r2, r3);
      if (k0 + 32 < I_) __builtin_prefetch(src + 32 * H_, 0, 0);
    }
    __syncthreads();

    v16h a[4], b[2];
#pragma unroll
    for (int mi = 0; mi < 4; ++mi) {
      v8h lo = *(const v8h*)&As[mWave + mi * 16 + lr][half * 8];
      v8h hi = *(const v8h*)&As[mWave + mi * 16 + lr][16 + half * 8];
#pragma unroll
      for (int j = 0; j < 8; ++j) { a[mi][j] = lo[j]; a[mi][8 + j] = hi[j]; }
    }
#pragma unroll
    for (int ni = 0; ni < 2; ++ni) {
      v8h l0 = *(const v8h*)&Bs[nWave + ni * 16 + lr][half * 8];
      v8h l1 = *(const v8h*)&Bs[nWave + ni * 16 + lr][16 + half * 8];
#pragma unroll
      for (int j = 0; j < 8; ++j) { b[ni][j] = l0[j]; b[ni][8 + j] = l1[j]; }
    }
#pragma unroll
    for (int mi = 0; mi < 4; ++mi)
#pragma unroll
      for (int ni = 0; ni < 2; ++ni)
        acc[mi][ni] = __builtin_amdgcn_wmma_f32_16x16x32_f16(false, a[mi], false, b[ni], (short)0, acc[mi][ni], false, false);
  }

  // scatter-add: top-1 routing => each token row is touched by exactly one expert
#pragma unroll
  for (int mi = 0; mi < 4; ++mi) {
    int lrow0 = mWave + mi * 16 + half * 8;
#pragma unroll
    for (int ni = 0; ni < 2; ++ni) {
      int col = nBase + nWave + ni * 16 + lr;
#pragma unroll
      for (int r = 0; r < 8; ++r) {
        int lrow = lrow0 + r;
        if (mBase + lrow < cnt) {
          int tok = rowTok[lrow];
          float* p = out + (size_t)tok * H_ + col;
          *p += acc[mi][ni][r];
        }
      }
    }
  }
}

// ---------------------------------------------------------------- launch ----
extern "C" void kernel_launch(void* const* d_in, const int* in_sizes, int n_in,
                              void* d_out, int out_size, void* d_ws, size_t ws_size,
                              hipStream_t stream) {
  const float* x   = (const float*)d_in[0];
  const float* rw  = (const float*)d_in[1];
  const float* wgu = (const float*)d_in[2];
  const float* wd  = (const float*)d_in[3];
  const float* sg  = (const float*)d_in[4];
  const float* su  = (const float*)d_in[5];
  const float* sd  = (const float*)d_in[6];
  float* out = (float*)d_out;
  float* scores_out = out + (size_t)T_ * H_;

  char* ws = (char*)d_ws;
  int*       hdr    = (int*)(ws + WS_HDR);
  int*       expert = (int*)(ws + WS_EXPERT);
  float*     score  = (float*)(ws + WS_SCORE);
  int*       perm   = (int*)(ws + WS_PERM);
  _Float16*  Sact   = (_Float16*)(ws + WS_SACT);
  _Float16*  Aact   = (_Float16*)(ws + WS_AACT);

  moe_init<<<1, 32, 0, stream>>>(hdr);
  moe_router<<<T_, 256, 0, stream>>>(x, rw, scores_out, expert, score, hdr);
  moe_scan<<<1, 1, 0, stream>>>(hdr);
  moe_assign<<<T_ / 256, 256, 0, stream>>>(expert, hdr, perm);

  gemm_gateup_shared<<<dim3(I_ / 64, T_ / 128), 256, 0, stream>>>(x, sg, su, Sact);
  gemm_down_shared<<<dim3(H_ / 128, T_ / 128), 256, 0, stream>>>(Sact, sd, out);

  gemm_gateup_routed<<<dim3(I_ / 64, T_ / 128, E_), 256, 0, stream>>>(x, wgu, Aact, hdr, perm, score);
  gemm_down_routed<<<dim3(H_ / 128, T_ / 128, E_), 256, 0, stream>>>(Aact, wd, out, hdr, perm);
}
